// GIN_4741643895039
// MI455X (gfx1250) — matrix operand order repaired and checked by reading, not compiled, not run on hardware
//
#include <hip/hip_runtime.h>

// ---------------------------------------------------------------------------
// GINE GNN forward for gfx1250 (MI455X): wave32 + v_wmma_f32_16x16x32_f16.
// Weights pre-swizzled once into WMMA B-operand f16 layout (global, L2-hot);
// A tiles staged in LDS as f16; pooling + agg-seed fused into GEMM2 epilogue.
// ---------------------------------------------------------------------------

typedef __attribute__((ext_vector_type(16))) _Float16 v16h;
typedef __attribute__((ext_vector_type(8)))  _Float16 v8h;
typedef __attribute__((ext_vector_type(8)))  float    v8f;

#define DEVINL __device__ __forceinline__

constexpr int N_NODES  = 100000;
constexpr int N_EDGES  = 1600000;
constexpr int DIMH     = 64;
constexpr int EMB_DIM  = 16;
constexpr int NGRAPHS  = 1000;
constexpr int GCOLS    = 3 * DIMH;           // 192
constexpr float BN_EPS = 1e-5f;

DEVINL void atomAddF(float* p, float v) {
  // relaxed, device-scope fp32 atomic -> native global_atomic_add_f32 (L2 RMW)
  __hip_atomic_fetch_add(p, v, __ATOMIC_RELAXED, __HIP_MEMORY_SCOPE_AGENT);
}

// ---------------------------------------------------------------------------
// WMMA 16x16 output tile. A from LDS (row-major f16, stride Kpad); B from
// global pre-swizzled layout Bsw[n * Kpad + k] (f16, zero-padded K).
// A-fragment (16-bit A 16x32), contiguous K to match B:
//   lane<16 : M=lane,    K = kk + 0..15
//   lane>=16: M=lane-16, K = kk + 16..31
// B-fragment (32x16): lane<16 -> K=kk+0..15, lane>=16 -> K=kk+16..31;
//   the 16 halves are 32 contiguous bytes -> 2x global_load_b128.
// ---------------------------------------------------------------------------
DEVINL v8f wmma_tiles(const _Float16* ldsA, const _Float16* __restrict__ Bsw,
                      int Kpad, int lane, int ncol) {
  v8f acc = {};
  const int m    = lane & 15;
  const int khiA = (lane < 16) ? 0 : 16;
  const int khiB = (lane < 16) ? 0 : 16;
  for (int kk = 0; kk < Kpad; kk += 32) {
    const _Float16* ap = &ldsA[m * Kpad + kk + khiA];
    v8h alo = *(const v8h*)(ap);            // 16B aligned
    v8h ahi = *(const v8h*)(ap + 8);
    v16h a;
#pragma unroll
    for (int i = 0; i < 8; ++i) { a[i] = alo[i]; a[i + 8] = ahi[i]; }
    v16h b = *(const v16h*)&Bsw[ncol * Kpad + kk + khiB];   // 32B aligned
    acc = __builtin_amdgcn_wmma_f32_16x16x32_f16(false, a, false, b,
                                                 (short)0, acc, false, false);
  }
  return acc;
}

// ---------------------------------------------------------------------------
// Elementwise / prep kernels
// ---------------------------------------------------------------------------
__global__ void k_zero(float* __restrict__ p, int n) {
  int t = blockIdx.x * blockDim.x + threadIdx.x;
  if (t < n) p[t] = 0.0f;
}

// Swizzle weight W[Kact x Nout] (f32, row-major) -> out[n * Kpad + k] (f16),
// zero-padded for k >= Kact. Tiny; runs once per weight per launch.
__global__ void k_prep_w(const float* __restrict__ W, _Float16* __restrict__ out,
                         int Kact, int Kpad, int Nout) {
  int t = blockIdx.x * blockDim.x + threadIdx.x;
  if (t >= Nout * Kpad) return;
  int n = t / Kpad, k = t - n * Kpad;
  out[t] = (k < Kact) ? (_Float16)W[k * Nout + n] : (_Float16)0.0f;
}

// h0[n,c] = (c < 16) ? emb[x[n], c] : 0; dual-write to h0 and agg seed.
__global__ void k_embed(const int* __restrict__ x, const float* __restrict__ emb,
                        float* __restrict__ h0, float* __restrict__ agg) {
  int t = blockIdx.x * blockDim.x + threadIdx.x;
  if (t >= N_NODES * DIMH) return;
  int n = t >> 6, c = t & 63;
  float v = (c < EMB_DIM) ? emb[x[n] * EMB_DIM + c] : 0.0f;
  h0[t]  = v;
  agg[t] = v;
}

// agg[dst] += relu(h[src] + edge_attr*We + be)   (agg pre-seeded with h)
__global__ void k_edge(const float* __restrict__ h, const int* __restrict__ src,
                       const int* __restrict__ dst, const float* __restrict__ eattr,
                       const float* __restrict__ We, const float* __restrict__ be,
                       float* __restrict__ agg, int din) {
  int t = blockIdx.x * blockDim.x + threadIdx.x;
  if (t >= N_EDGES * din) return;
  int e = t / din, c = t - e * din;
  float m = h[src[e] * DIMH + c] + eattr[e] * We[c] + be[c];
  if (m > 0.0f) atomAddF(&agg[dst[e] * DIMH + c], m);
}

// ---------------------------------------------------------------------------
// GEMM1: Z = A @ W1 + b1 (in-place on A; A staged to LDS first) plus
// per-column sum / sumsq for batch-norm. A: [N,64] (cols >= Kact are zero).
// Block = 128 threads (4 waves); one 16-row stripe; wave w -> cols 16w..16w+15.
// N % 16 == 0 -> no row guards.
// ---------------------------------------------------------------------------
__global__ void __launch_bounds__(128)
k_linear_stats(const float* __restrict__ A, const _Float16* __restrict__ Bsw,
               const float* __restrict__ bias, float* __restrict__ Z,
               float* __restrict__ stats) {
  __shared__ __align__(32) _Float16 ldsA[16 * DIMH];
  const int wave = threadIdx.x >> 5, lane = threadIdx.x & 31;
  const int row0 = blockIdx.x * 16;

  const float4* A4 = (const float4*)(A + row0 * DIMH);
  for (int idx = threadIdx.x; idx < 256; idx += 128) {     // 16*64/4
    float4 v = A4[idx];
    int base = idx * 4;
    ldsA[base + 0] = (_Float16)v.x;
    ldsA[base + 1] = (_Float16)v.y;
    ldsA[base + 2] = (_Float16)v.z;
    ldsA[base + 3] = (_Float16)v.w;
  }
  __syncthreads();

  const int col = wave * 16 + (lane & 15);
  v8f acc = wmma_tiles(ldsA, Bsw, DIMH, lane, col);

  const int rbit = (lane < 16) ? 0 : 8;
  const float b  = bias[col];
  float s = 0.0f, sq = 0.0f;
#pragma unroll
  for (int j = 0; j < 8; ++j) {
    float z = acc[j] + b;
    Z[(row0 + j + rbit) * DIMH + col] = z;
    s  += z;
    sq += z * z;
  }
  atomAddF(&stats[col], s);
  atomAddF(&stats[DIMH + col], sq);
}

// ---------------------------------------------------------------------------
// GEMM2 + fused epilogue:
//   hout = relu( relu(BN(Zin)) @ W2 + b2 )
//   aggNext := hout                (agg seed for next layer; Zin aliased OK)
//   g[batch[row], colOff + col] += hout   (graph pooling, f32 atomics)
// ---------------------------------------------------------------------------
__global__ void __launch_bounds__(128)
k_bn_linear_relu_pool(const float* __restrict__ Zin, const float* __restrict__ stats,
                      const float* __restrict__ gamma, const float* __restrict__ beta,
                      const _Float16* __restrict__ Bsw, const float* __restrict__ bias,
                      float* __restrict__ Hout, float* __restrict__ aggNext,
                      const int* __restrict__ batch, float* __restrict__ g,
                      int colOff) {
  __shared__ __align__(32) _Float16 ldsA[16 * DIMH];
  __shared__ float sc[DIMH], sh[DIMH];
  __shared__ int sbatch[16];
  const int wave = threadIdx.x >> 5, lane = threadIdx.x & 31;
  const int row0 = blockIdx.x * 16;

  if (threadIdx.x < DIMH) {
    const float invM = 1.0f / (float)N_NODES;
    float mu  = stats[threadIdx.x] * invM;
    float var = stats[DIMH + threadIdx.x] * invM - mu * mu;
    float is  = rsqrtf(var + BN_EPS);
    float gsc = gamma[threadIdx.x] * is;
    sc[threadIdx.x] = gsc;
    sh[threadIdx.x] = beta[threadIdx.x] - mu * gsc;
  }
  if (threadIdx.x < 16) sbatch[threadIdx.x] = batch[row0 + threadIdx.x];
  __syncthreads();

  const float4* Z4 = (const float4*)(Zin + row0 * DIMH);
  for (int idx = threadIdx.x; idx < 256; idx += 128) {
    float4 v = Z4[idx];
    int base = idx * 4;
    int c = base & 63;
#pragma unroll
    for (int j = 0; j < 4; ++j) {
      float f = (&v.x)[j] * sc[c + j] + sh[c + j];
      ldsA[base + j] = (_Float16)fmaxf(f, 0.0f);           // relu(BN(z))
    }
  }
  __syncthreads();

  const int col = wave * 16 + (lane & 15);
  v8f acc = wmma_tiles(ldsA, Bsw, DIMH, lane, col);

  const int rbit = (lane < 16) ? 0 : 8;
  const float b  = bias[col];
#pragma unroll
  for (int j = 0; j < 8; ++j) {
    int row = row0 + j + rbit;
    float v = fmaxf(acc[j] + b, 0.0f);
    Hout[row * DIMH + col]    = v;
    aggNext[row * DIMH + col] = v;                          // seed next agg
    atomAddF(&g[sbatch[j + rbit] * GCOLS + colOff + col], v);
  }
}

// ---------------------------------------------------------------------------
// Head GEMM: Z[M,Nout] = act(A[M,K] @ W + b). K in {64,192}; A dense (lda==K).
// grid.y tiles Nout in chunks of 64.
// ---------------------------------------------------------------------------
__global__ void __launch_bounds__(128)
k_head_linear(const float* __restrict__ A, int M, int K,
              const _Float16* __restrict__ Bsw, const float* __restrict__ bias,
              int Nout, float* __restrict__ Z, int doRelu) {
  __shared__ __align__(32) _Float16 ldsA[16 * GCOLS];      // max K = 192
  const int wave = threadIdx.x >> 5, lane = threadIdx.x & 31;
  const int row0  = blockIdx.x * 16;
  const int nbase = blockIdx.y * 64;

  const int nA4 = 16 * K / 4;
  for (int idx = threadIdx.x; idx < nA4; idx += 128) {
    int base = idx * 4;
    int r = base / K, c = base - r * K;
    int rr = row0 + r;
    float4 v = make_float4(0.f, 0.f, 0.f, 0.f);
    if (rr < M) v = *(const float4*)(A + rr * K + c);
    ldsA[base + 0] = (_Float16)v.x;
    ldsA[base + 1] = (_Float16)v.y;
    ldsA[base + 2] = (_Float16)v.z;
    ldsA[base + 3] = (_Float16)v.w;
  }
  __syncthreads();

  const int col = nbase + wave * 16 + (lane & 15);
  v8f acc = wmma_tiles(ldsA, Bsw, K, lane, col);

  const int rbit = (lane < 16) ? 0 : 8;
  const float b  = bias[col];
#pragma unroll
  for (int j = 0; j < 8; ++j) {
    int row = row0 + j + rbit;
    if (row < M) {
      float v = acc[j] + b;
      if (doRelu) v = fmaxf(v, 0.0f);
      Z[row * Nout + col] = v;
    }
  }
}

// ---------------------------------------------------------------------------
// Host-side orchestration
// ---------------------------------------------------------------------------
extern "C" void kernel_launch(void* const* d_in, const int* in_sizes, int n_in,
                              void* d_out, int out_size, void* d_ws, size_t ws_size,
                              hipStream_t stream) {
  (void)in_sizes; (void)n_in; (void)out_size; (void)ws_size;

  // inputs in setup_inputs() insertion order
  const int*   x     = (const int*)  d_in[0];
  const int*   eidx  = (const int*)  d_in[1];             // [2, E]
  const float* eattr = (const float*)d_in[2];
  const int*   batch = (const int*)  d_in[3];
  const float* emb   = (const float*)d_in[4];
  const int*   src   = eidx;
  const int*   dst   = eidx + N_EDGES;
  const int convBase[3] = {5, 13, 21};  // We, be, W1, b1, gamma, beta, W2, b2
  const int headBase[2] = {29, 33};     // W1, b1, W2, b2

  // workspace layout
  float* bufA  = (float*)d_ws;                            // [N, 64]
  float* bufB  = bufA  + (size_t)N_NODES * DIMH;          // [N, 64]
  float* zz    = bufB  + (size_t)N_NODES * DIMH;          // agg / z, [N, 64]
  float* stats = zz    + (size_t)N_NODES * DIMH;          // [128]
  float* g     = stats + 2 * DIMH;                        // [NGRAPHS, 192]
  float* t     = g     + (size_t)NGRAPHS * GCOLS;         // [NGRAPHS, 64]
  _Float16* wf = (_Float16*)(t + (size_t)NGRAPHS * DIMH); // f16 swizzled weights
  _Float16* convW[6];                                     // W1,W2 per layer
  for (int i = 0; i < 6; ++i) convW[i] = wf + (size_t)i * 4096;
  _Float16* headW1[2] = { wf + 24576, wf + 24576 + 12288 };
  _Float16* headW2[2] = { wf + 49152, wf + 49152 + 16384 };

  const int NH       = N_NODES * DIMH;
  const int gridNH   = (NH + 255) / 256;
  const int rowTiles = N_NODES / 16;                      // 6250

  // --- weight pre-swizzle (once per launch) ---
  for (int L = 0; L < 3; ++L) {
    const int pb  = convBase[L];
    const int din = (L == 0) ? EMB_DIM : DIMH;
    k_prep_w<<<(DIMH * DIMH + 255) / 256, 256, 0, stream>>>(
        (const float*)d_in[pb + 2], convW[2 * L + 0], din, DIMH, DIMH);
    k_prep_w<<<(DIMH * DIMH + 255) / 256, 256, 0, stream>>>(
        (const float*)d_in[pb + 6], convW[2 * L + 1], DIMH, DIMH, DIMH);
  }
  for (int h = 0; h < 2; ++h) {
    const int pb = headBase[h];
    k_prep_w<<<(DIMH * GCOLS + 255) / 256, 256, 0, stream>>>(
        (const float*)d_in[pb + 0], headW1[h], GCOLS, GCOLS, DIMH);
    k_prep_w<<<(256 * DIMH + 255) / 256, 256, 0, stream>>>(
        (const float*)d_in[pb + 2], headW2[h], DIMH, DIMH, 256);
  }

  k_embed<<<gridNH, 256, 0, stream>>>(x, emb, bufA, zz);  // h0 + agg seed
  k_zero<<<(NGRAPHS * GCOLS + 255) / 256, 256, 0, stream>>>(g, NGRAPHS * GCOLS);

  for (int L = 0; L < 3; ++L) {
    const float* hin  = (L == 1) ? bufB : bufA;
    float*       hout = (L == 1) ? bufA : bufB;
    const int din = (L == 0) ? EMB_DIM : DIMH;
    const int pb  = convBase[L];
    const float* We    = (const float*)d_in[pb + 0];
    const float* be    = (const float*)d_in[pb + 1];
    const float* b1    = (const float*)d_in[pb + 3];
    const float* gamma = (const float*)d_in[pb + 4];
    const float* beta  = (const float*)d_in[pb + 5];
    const float* b2    = (const float*)d_in[pb + 7];

    k_zero<<<1, 128, 0, stream>>>(stats, 2 * DIMH);

    const int ework = N_EDGES * din;
    k_edge<<<(ework + 255) / 256, 256, 0, stream>>>(hin, src, dst, eattr,
                                                    We, be, zz, din);

    k_linear_stats<<<rowTiles, 128, 0, stream>>>(zz, convW[2 * L], b1, zz, stats);
    k_bn_linear_relu_pool<<<rowTiles, 128, 0, stream>>>(
        zz, stats, gamma, beta, convW[2 * L + 1], b2,
        hout, zz, batch, g, L * DIMH);
  }

  // heads: relu(g @ W1 + b1) @ W2 + b2  -> [1000, 256] each, concat in d_out
  const int gTiles = (NGRAPHS + 15) / 16;                 // 63
  for (int hsel = 0; hsel < 2; ++hsel) {
    const int pb = headBase[hsel];
    const float* b1 = (const float*)d_in[pb + 1];
    const float* b2 = (const float*)d_in[pb + 3];
    float* outp = (float*)d_out + (size_t)hsel * NGRAPHS * 256;

    k_head_linear<<<dim3(gTiles, 1), 128, 0, stream>>>(g, NGRAPHS, GCOLS,
                                                       headW1[hsel], b1, DIMH, t, 1);
    k_head_linear<<<dim3(gTiles, 4), 128, 0, stream>>>(t, NGRAPHS, DIMH,
                                                       headW2[hsel], b2, 256, outp, 0);
  }
}
